// LSTMCell_65120294142269
// MI455X (gfx1250) — matrix operand order repaired and checked by reading, not compiled, not run on hardware
//
#include <hip/hip_runtime.h>
#include <hip/hip_bf16.h>
#include <math.h>

// ---------------------------------------------------------------------------
// Problem constants (from reference): B=8192, I=512, H=1024
//   gates(8192,4096) = [x|h](8192,1536) @ [Wi;Wh](1536,4096) + bh
// ---------------------------------------------------------------------------
#define MB_   8192
#define KTOT  1536
#define NTOT  4096
#define HDIM  1024

#define BM 128
#define BN 128
#define BK 32

typedef __attribute__((ext_vector_type(16))) __bf16 v16bf;
typedef __attribute__((ext_vector_type(8)))  float  v8f;
typedef __attribute__((ext_vector_type(8)))  unsigned short u16x8;

union Frag {
    u16x8 h[2];
    v16bf v;
};

// ---------------------------------------------------------------------------
// CDNA5 async global->LDS copy (16 bytes per lane), ASYNCcnt-tracked.
// Builtin signature (from clang diagnostic):
//   void __builtin_amdgcn_global_load_async_to_lds_b128(
//       __device__(AS1) int4_vs* src, __shared__(AS3) int4_vs* dst,
//       imm int offset, imm int cpol)
// ---------------------------------------------------------------------------
typedef int v4i_vs __attribute__((vector_size(4 * sizeof(int))));
#define AS_GLOBAL __attribute__((address_space(1)))
#define AS_SHARED __attribute__((address_space(3)))

#if __has_builtin(__builtin_amdgcn_global_load_async_to_lds_b128)
#define ASYNC_CP16(gsrc, ldst)                                                \
    __builtin_amdgcn_global_load_async_to_lds_b128(                           \
        (AS_GLOBAL v4i_vs*)(gsrc), (AS_SHARED v4i_vs*)(ldst), 0, 0)
#else
#define ASYNC_CP16(gsrc, ldst)                                                \
    asm volatile("global_load_async_to_lds_b128 %0, %1, off"                  \
                 :: "v"((unsigned)(unsigned long long)(const void*)(ldst)),   \
                    "v"((unsigned long long)(const void*)(gsrc))              \
                 : "memory")
#endif

#if __has_builtin(__builtin_amdgcn_s_wait_asynccnt)
#define WAIT_ASYNC() __builtin_amdgcn_s_wait_asynccnt(0)
#else
#define WAIT_ASYNC() asm volatile("s_wait_asynccnt 0x0" ::: "memory")
#endif

__device__ __forceinline__ unsigned short f2bf(float f) {
    unsigned u = __builtin_bit_cast(unsigned, f);
    unsigned r = u + 0x7fffu + ((u >> 16) & 1u);   // round-to-nearest-even
    return (unsigned short)(r >> 16);
}

// ---------------------------------------------------------------------------
// Transcode A = [x | h] (row-major 8192 x 1536) fp32 -> bf16
// ---------------------------------------------------------------------------
__global__ __launch_bounds__(256)
void convA_kernel(const float* __restrict__ x, const float* __restrict__ h,
                  unsigned short* __restrict__ A)
{
    size_t i = (size_t)blockIdx.x * blockDim.x + threadIdx.x;
    size_t e = i * 4;                       // 4 consecutive k per thread
    int b = (int)(e / KTOT);
    int k = (int)(e % KTOT);
    const float* src = (k < 512) ? (x + (size_t)b * 512 + k)
                                 : (h + (size_t)b * 1024 + (k - 512));
    float4 v = *(const float4*)src;
    ushort4 o;
    o.x = f2bf(v.x); o.y = f2bf(v.y); o.z = f2bf(v.z); o.w = f2bf(v.w);
    *(ushort4*)(A + e) = o;
}

// ---------------------------------------------------------------------------
// Transcode Bt[n][k] = (k<512 ? Wi[k][n] : Wh[k-512][n])  (4096 x 1536 bf16)
// ---------------------------------------------------------------------------
__global__ __launch_bounds__(256)
void convB_kernel(const float* __restrict__ Wi, const float* __restrict__ Wh,
                  unsigned short* __restrict__ Bt)
{
    size_t i = (size_t)blockIdx.x * blockDim.x + threadIdx.x;
    size_t e = i * 4;
    int n = (int)(e / KTOT);
    int k = (int)(e % KTOT);
    ushort4 o;
    float v0, v1, v2, v3;
    if (k < 512) {
        v0 = Wi[(size_t)(k + 0) * NTOT + n];
        v1 = Wi[(size_t)(k + 1) * NTOT + n];
        v2 = Wi[(size_t)(k + 2) * NTOT + n];
        v3 = Wi[(size_t)(k + 3) * NTOT + n];
    } else {
        int kk = k - 512;
        v0 = Wh[(size_t)(kk + 0) * NTOT + n];
        v1 = Wh[(size_t)(kk + 1) * NTOT + n];
        v2 = Wh[(size_t)(kk + 2) * NTOT + n];
        v3 = Wh[(size_t)(kk + 3) * NTOT + n];
    }
    o.x = f2bf(v0); o.y = f2bf(v1); o.z = f2bf(v2); o.w = f2bf(v3);
    *(ushort4*)(Bt + e) = o;
}

// ---------------------------------------------------------------------------
// bf16 WMMA GEMM: gates = A(8192x1536) @ Bt^T(1536x4096) + bh
// block 128x128, 256 threads = 8 wave32 (2x4), wave tile 64x32 = 4x2 WMMAs
// Global->LDS staging uses async copies (ASYNCcnt), double-buffered LDS,
// single barrier per K-iteration.
// ---------------------------------------------------------------------------
__global__ __launch_bounds__(256)
void gemm_bf16_kernel(const unsigned short* __restrict__ A,
                      const unsigned short* __restrict__ Bt,
                      const float* __restrict__ bh,
                      float* __restrict__ gates)
{
    __shared__ unsigned short As[2][BM][BK];
    __shared__ unsigned short Bs[2][BN][BK];

    const int tid   = threadIdx.x;
    const int lane  = tid & 31;
    const int wave  = tid >> 5;
    const int waveM = wave >> 2;      // 0..1  (rows, 64 each)
    const int waveN = wave & 3;       // 0..3  (cols, 32 each)
    const int half  = lane >> 4;      // lane half -> K chunk select
    const int lrow  = lane & 15;      // M (or N) index within tile

    const int rowBase = blockIdx.y * BM;
    const int colBase = blockIdx.x * BN;

    // global->LDS staging: each thread owns one 32B chunk of each tile
    const int ldRow = tid >> 1;           // 0..127
    const int ldOff = (tid & 1) * 16;     // element offset within 32-wide row

    const unsigned short* Aptr = A  + (size_t)(rowBase + ldRow) * KTOT + ldOff;
    const unsigned short* Bptr = Bt + (size_t)(colBase + ldRow) * KTOT + ldOff;

    // prologue: stage tile 0 asynchronously
    ASYNC_CP16(Aptr,     &As[0][ldRow][ldOff]);
    ASYNC_CP16(Aptr + 8, &As[0][ldRow][ldOff + 8]);
    ASYNC_CP16(Bptr,     &Bs[0][ldRow][ldOff]);
    ASYNC_CP16(Bptr + 8, &Bs[0][ldRow][ldOff + 8]);
    WAIT_ASYNC();
    __syncthreads();

    v8f acc[4][2] = {};

    const int KITERS = KTOT / BK;   // 48
    for (int kk = 0; kk < KITERS; ++kk) {
        const int buf = kk & 1;
        const int nb  = buf ^ 1;

        if (kk + 1 < KITERS) {
            const unsigned short* ap = Aptr + (size_t)(kk + 1) * BK;
            const unsigned short* bp = Bptr + (size_t)(kk + 1) * BK;
            ASYNC_CP16(ap,     &As[nb][ldRow][ldOff]);
            ASYNC_CP16(ap + 8, &As[nb][ldRow][ldOff + 8]);
            ASYNC_CP16(bp,     &Bs[nb][ldRow][ldOff]);
            ASYNC_CP16(bp + 8, &Bs[nb][ldRow][ldOff + 8]);
            if (kk + 2 < KITERS) {
                __builtin_prefetch(Aptr + (size_t)(kk + 2) * BK, 0, 1);
                __builtin_prefetch(Bptr + (size_t)(kk + 2) * BK, 0, 1);
            }
        }

        // load fragments from LDS (documented 16-bit A layout: K chunks
        // {half*8 .. +7} and {half*8+16 .. +23}; B fragment: N=lane%16,
        // same K chunks as A)
        Frag afr[4], bfr[2];
#pragma unroll
        for (int t = 0; t < 4; ++t) {
            const int r = waveM * 64 + t * 16 + lrow;
            afr[t].h[0] = *(const u16x8*)&As[buf][r][half * 8];
            afr[t].h[1] = *(const u16x8*)&As[buf][r][half * 8 + 16];
        }
#pragma unroll
        for (int u = 0; u < 2; ++u) {
            const int n = waveN * 32 + u * 16 + lrow;
            bfr[u].h[0] = *(const u16x8*)&Bs[buf][n][half * 8];
            bfr[u].h[1] = *(const u16x8*)&Bs[buf][n][half * 8 + 16];
        }

#pragma unroll
        for (int t = 0; t < 4; ++t)
#pragma unroll
            for (int u = 0; u < 2; ++u)
                acc[t][u] = __builtin_amdgcn_wmma_f32_16x16x32_bf16(
                    /*neg_a=*/false, afr[t].v,
                    /*neg_b=*/false, bfr[u].v,
                    /*c_mod=*/(short)0, acc[t][u],
                    /*reuse_a=*/false, /*reuse_b=*/false);

        if (kk + 1 < KITERS) {
            WAIT_ASYNC();          // this wave's async writes to buf^1 landed
        }
        __syncthreads();           // all waves: reads of buf done, nb visible
    }

    // epilogue: C layout — VGPR vr, lane<16: (M=vr, N=lane); lane>=16: (M=vr+8)
    float bhv[2];
#pragma unroll
    for (int u = 0; u < 2; ++u)
        bhv[u] = bh[colBase + waveN * 32 + u * 16 + lrow];

#pragma unroll
    for (int t = 0; t < 4; ++t) {
#pragma unroll
        for (int u = 0; u < 2; ++u) {
            const int n = colBase + waveN * 32 + u * 16 + lrow;
#pragma unroll
            for (int vr = 0; vr < 8; ++vr) {
                const int m = rowBase + waveM * 64 + t * 16 + half * 8 + vr;
                gates[(size_t)m * NTOT + n] = acc[t][u][vr] + bhv[u];
            }
        }
    }
}

// ---------------------------------------------------------------------------
// Per-row LayerNorm (4 gates x 1024) + LSTM pointwise.
// One block per row; 256 threads, 4 columns/thread.
// ---------------------------------------------------------------------------
__device__ __forceinline__ float sigmoidf_(float v) { return 1.0f / (1.0f + __expf(-v)); }

__global__ __launch_bounds__(256)
void ln_lstm_kernel(const float* __restrict__ gates,
                    const float* __restrict__ c,
                    const float* __restrict__ gamma,
                    const float* __restrict__ beta,
                    float* __restrict__ out)
{
    __shared__ float redS[8];
    __shared__ float redQ[8];

    const int b    = blockIdx.x;
    const int tid  = threadIdx.x;
    const int lane = tid & 31;
    const int wv   = tid >> 5;

    float y[4][4];

#pragma unroll
    for (int g = 0; g < 4; ++g) {
        const float4 v = *(const float4*)(gates + (size_t)b * NTOT + g * HDIM + tid * 4);
        float s = v.x + v.y + v.z + v.w;
        float q = v.x * v.x + v.y * v.y + v.z * v.z + v.w * v.w;
#pragma unroll
        for (int off = 16; off > 0; off >>= 1) {
            s += __shfl_xor(s, off, 32);
            q += __shfl_xor(q, off, 32);
        }
        __syncthreads();                 // previous gate's reads of red* done
        if (lane == 0) { redS[wv] = s; redQ[wv] = q; }
        __syncthreads();
        if (tid == 0) {
            float ss = 0.f, qq = 0.f;
#pragma unroll
            for (int w = 0; w < 8; ++w) { ss += redS[w]; qq += redQ[w]; }
            redS[0] = ss; redQ[0] = qq;
        }
        __syncthreads();
        const float mu   = redS[0] * (1.0f / HDIM);
        const float var  = redQ[0] * (1.0f / HDIM) - mu * mu;
        const float rstd = rsqrtf(var + 1e-5f);

        const float4 gm = *(const float4*)(gamma + g * HDIM + tid * 4);
        const float4 bt = *(const float4*)(beta  + g * HDIM + tid * 4);
        y[g][0] = (v.x - mu) * rstd * gm.x + bt.x;
        y[g][1] = (v.y - mu) * rstd * gm.y + bt.y;
        y[g][2] = (v.z - mu) * rstd * gm.z + bt.z;
        y[g][3] = (v.w - mu) * rstd * gm.w + bt.w;
    }

    const float4 cv = *(const float4*)(c + (size_t)b * HDIM + tid * 4);
    const float cvals[4] = { cv.x, cv.y, cv.z, cv.w };

    float4 hnew, cnew;
    float* hn = &hnew.x;
    float* cn = &cnew.x;
#pragma unroll
    for (int j = 0; j < 4; ++j) {
        const float iG = sigmoidf_(y[0][j]);
        const float fG = sigmoidf_(y[1][j]);
        const float gG = tanhf(y[2][j]);
        const float oG = sigmoidf_(y[3][j]);
        const float cc = fG * cvals[j] + iG * gG;
        cn[j] = cc;
        hn[j] = oG * tanhf(cc);
    }

    const size_t col = (size_t)b * HDIM + tid * 4;
    *(float4*)(out + col)                          = hnew;  // h_new
    *(float4*)(out + (size_t)MB_ * HDIM + col)     = cnew;  // c_new
}

// ---------------------------------------------------------------------------
// Launch
// ---------------------------------------------------------------------------
extern "C" void kernel_launch(void* const* d_in, const int* in_sizes, int n_in,
                              void* d_out, int out_size, void* d_ws, size_t ws_size,
                              hipStream_t stream)
{
    const float* x     = (const float*)d_in[0];
    const float* h     = (const float*)d_in[1];
    const float* c     = (const float*)d_in[2];
    const float* Wi    = (const float*)d_in[3];
    const float* Wh    = (const float*)d_in[4];
    const float* bh    = (const float*)d_in[5];
    const float* gamma = (const float*)d_in[6];
    const float* beta  = (const float*)d_in[7];
    float* out = (float*)d_out;

    unsigned short* Abf  = (unsigned short*)d_ws;                      // 8192*1536 bf16
    unsigned short* Btbf = Abf + (size_t)MB_ * KTOT;                   // 4096*1536 bf16
    float*          gts  = (float*)(Btbf + (size_t)NTOT * KTOT);       // 8192*4096 f32

    convA_kernel<<<(MB_ * KTOT) / (256 * 4), 256, 0, stream>>>(x, h, Abf);
    convB_kernel<<<(NTOT * KTOT) / (256 * 4), 256, 0, stream>>>(Wi, Wh, Btbf);

    dim3 grid(NTOT / BN, MB_ / BM);   // (32, 64)
    gemm_bf16_kernel<<<grid, 256, 0, stream>>>(Abf, Btbf, bh, gts);

    ln_lstm_kernel<<<MB_, 256, 0, stream>>>(gts, c, gamma, beta, out);
}
